// TemporalDifferentModuleMSDeformAttn_34196529610804
// MI455X (gfx1250) — matrix operand order repaired and checked by reading, not compiled, not run on hardware
//
#include <hip/hip_runtime.h>
#include <hip/hip_bf16.h>

typedef __attribute__((ext_vector_type(16))) __bf16 v16bf;
typedef __attribute__((ext_vector_type(8)))  __bf16 v8bf;
typedef __attribute__((ext_vector_type(8)))  float  v8f;
typedef __attribute__((ext_vector_type(4)))  float  v4f;

#define LQ      24480
#define DMODEL  256
#define NHEADS  8
#define NLEVELS 4
#define DHEAD   32
#define KDIM    256

// ---------------------------------------------------------------------------
// One-time weight prep: Wt[n*K + k] = bf16(W[k*N + n])  (transpose + convert)
// grid.x = N, block = K threads.
// ---------------------------------------------------------------------------
__global__ void transpose_convert_w(const float* __restrict__ W,
                                    __bf16* __restrict__ Wt, int N)
{
    const int n = blockIdx.x;
    const int k = threadIdx.x;            // 0..255 (K = 256)
    Wt[(size_t)n * KDIM + k] = (__bf16)W[(size_t)k * N + n];
}

// ---------------------------------------------------------------------------
// GEMM: C(M,N) = A(M,K) @ W(K,N) + bias(N); A fp32 row-major, W pre-transposed
// bf16 (N x K row-major), C fp32. One wave32 per 16x16 tile, 8 waves/block.
// M%16==0, N%16==0, K==256 for all calls here.
// ---------------------------------------------------------------------------
__global__ __launch_bounds__(256) void gemm_wmma_bf16(
    const float*  __restrict__ A, const __bf16* __restrict__ Wt,
    const float*  __restrict__ bias, float* __restrict__ C,
    int M, int N)
{
    const int lane = threadIdx.x;           // 0..31
    const int wave = threadIdx.y;           // 0..7
    const int tilesN = N >> 4;
    const int tiles  = (M >> 4) * tilesN;
    const int tile   = blockIdx.x * 8 + wave;
    if (tile >= tiles) return;
    const int tm = tile / tilesN;
    const int tn = tile % tilesN;

    const int hi  = lane >> 4;              // 0: lanes 0-15, 1: lanes 16-31
    const int l16 = lane & 15;

    const float*  arow = A  + (size_t)(tm * 16 + l16) * (size_t)KDIM;  // row M
    const int     ncol = tn * 16 + l16;                                // col N
    const __bf16* brow = Wt + (size_t)ncol * (size_t)KDIM;             // col as row

    v8f acc = {0.f, 0.f, 0.f, 0.f, 0.f, 0.f, 0.f, 0.f};

    #pragma unroll 2
    for (int k0 = 0; k0 < KDIM; k0 += 32) {
        // ---- A 16x32 bf16 fragment (ISA layout) ----
        //  lanes 0-15: K = k0+0..7, k0+16..23 ; lanes 16-31: +8
        const int ka = k0 + hi * 8;                   // multiple of 8 -> 32B aligned
        const v4f a0 = ((const v4f*)(arow + ka))[0];
        const v4f a1 = ((const v4f*)(arow + ka))[1];
        const v4f a2 = ((const v4f*)(arow + ka + 16))[0];
        const v4f a3 = ((const v4f*)(arow + ka + 16))[1];
        v16bf a;
        #pragma unroll
        for (int i = 0; i < 4; ++i) {
            a[i]      = (__bf16)a0[i];
            a[i + 4]  = (__bf16)a1[i];
            a[i + 8]  = (__bf16)a2[i];
            a[i + 12] = (__bf16)a3[i];
        }
        // ---- B 32x16 bf16 fragment: contiguous in transposed weights ----
        //  lanes 0-15: K = k0+0..7, k0+16..23 ; lanes 16-31: +8
        const int kb = k0 + hi * 8;                   // multiple of 8 -> 16B aligned
        union { v16bf v; v8bf h[2]; } bu;
        bu.h[0] = ((const v8bf*)(brow + kb))[0];
        bu.h[1] = ((const v8bf*)(brow + kb + 16))[0];

        acc = __builtin_amdgcn_wmma_f32_16x16x32_bf16(
            /*neg_a=*/false, a, /*neg_b=*/false, bu.v,
            /*c_mod=*/(short)0, acc, /*reuse_a=*/false, /*reuse_b=*/false);
    }

    // C layout: VGPR r -> M = r (lanes 0-15) or r+8 (lanes 16-31), N = l16
    const float bn = bias[ncol];
    float* cbase = C + (size_t)(tm * 16) * (size_t)N + (size_t)ncol;
    #pragma unroll
    for (int r = 0; r < 8; ++r) {
        const int m = r + hi * 8;
        cbase[(size_t)m * (size_t)N] = acc[r] + bn;
    }
}

// ---------------------------------------------------------------------------
// Deformable sampling: one block per query, one wave per head.
// Phase 1: lane = point (L*P = 32); joint softmax over all 32 logits.
// Phase 2: lane = channel (DHEAD = 32); coalesced 128B corner gathers.
// ---------------------------------------------------------------------------
__global__ __launch_bounds__(256) void sample_kernel(
    const float* __restrict__ value,   // (LEN_IN, 256)
    const float* __restrict__ caw,     // (Lq, 128)  H x (L*4)
    const float* __restrict__ taw,     // (Lq, 128)  H x (L*4)
    const float* __restrict__ cso,     // (Lq, 256)  H x L x 4 x 2
    const float* __restrict__ tso,     // (Lq, 256)  H x L x 2 x 2 x 2
    const float* __restrict__ refp,    // (Lq, L, 2)
    const float* __restrict__ toff,    // (Lq, L, 2, 2)
    float* __restrict__ outh)          // (Lq, 256)
{
    const int q    = blockIdx.x;
    const int h    = threadIdx.y;      // head
    const int lane = threadIdx.x;      // 0..31

    // ---- Phase 1: this lane owns point (l, p) ----
    const int l = lane >> 3;           // level 0..3
    const int p = lane & 7;            // 0-3 current, 4-7 temporal

    const float Wl  = (float)(64 >> l);
    const float HtL = (float)(288 >> l);    // T*H = 6*(48>>l)

    const float rx = refp[(q * NLEVELS + l) * 2 + 0];
    const float ry = refp[(q * NLEVELS + l) * 2 + 1];

    float logit, lx, ly;
    if (p < 4) {
        const int c = p;
        logit = caw[q * 128 + h * 16 + l * 4 + c];
        const int o = q * 256 + ((h * NLEVELS + l) * 4 + c) * 2;
        lx = rx + cso[o + 0] / Wl;
        ly = ry + cso[o + 1] / HtL;
    } else {
        const int j4 = p - 4;          // t*2 + j
        const int t  = j4 >> 1;
        const int j  = j4 & 1;
        logit = taw[q * 128 + h * 16 + l * 4 + j4];
        const int o  = q * 256 + (((h * NLEVELS + l) * 2 + t) * 2 + j) * 2;
        const int to = q * 16 + (l * 2 + t) * 2;
        lx = rx + toff[to + 0] + tso[o + 0] / Wl;
        ly = ry + toff[to + 1] + tso[o + 1] / HtL;
    }

    // Joint softmax over all 32 points of this (q, h)
    float mx = logit;
    #pragma unroll
    for (int s = 16; s > 0; s >>= 1) mx = fmaxf(mx, __shfl_xor(mx, s, 32));
    const float e = __expf(logit - mx);
    float ssum = e;
    #pragma unroll
    for (int s = 16; s > 0; s >>= 1) ssum += __shfl_xor(ssum, s, 32);
    const float w = e / ssum;

    // ---- Phase 2: this lane owns channel d = lane ----
    const float* vh = value + (size_t)h * DHEAD + lane;
    float acc = 0.f;
    for (int pt = 0; pt < 32; ++pt) {
        const float plx = __shfl(lx, pt, 32);
        const float ply = __shfl(ly, pt, 32);
        const float pw  = __shfl(w,  pt, 32);
        const int pl = pt >> 3;
        const int Wp = 64 >> pl;
        const int Hp = 288 >> pl;
        const int st = (pl == 0) ? 0 : (pl == 1) ? 18432 : (pl == 2) ? 23040 : 24192;

        const float x = plx * (float)Wp - 0.5f;
        const float y = ply * (float)Hp - 0.5f;
        const int x0 = (int)floorf(x);
        const int y0 = (int)floorf(y);
        const float fx = x - (float)x0;
        const float fy = y - (float)y0;

        float g = 0.f;   // zero padding outside the grid
        {
            const int yy = y0, xx = x0;
            if (yy >= 0 && yy < Hp && xx >= 0 && xx < Wp)
                g += (1.f - fx) * (1.f - fy) * vh[(size_t)(st + yy * Wp + xx) * DMODEL];
        }
        {
            const int yy = y0, xx = x0 + 1;
            if (yy >= 0 && yy < Hp && xx >= 0 && xx < Wp)
                g += fx * (1.f - fy) * vh[(size_t)(st + yy * Wp + xx) * DMODEL];
        }
        {
            const int yy = y0 + 1, xx = x0;
            if (yy >= 0 && yy < Hp && xx >= 0 && xx < Wp)
                g += (1.f - fx) * fy * vh[(size_t)(st + yy * Wp + xx) * DMODEL];
        }
        {
            const int yy = y0 + 1, xx = x0 + 1;
            if (yy >= 0 && yy < Hp && xx >= 0 && xx < Wp)
                g += fx * fy * vh[(size_t)(st + yy * Wp + xx) * DMODEL];
        }
        acc += pw * g;
    }
    outh[q * DMODEL + h * DHEAD + lane] = acc;
}

// ---------------------------------------------------------------------------
extern "C" void kernel_launch(void* const* d_in, const int* in_sizes, int n_in,
                              void* d_out, int out_size, void* d_ws, size_t ws_size,
                              hipStream_t stream) {
    const float* query   = (const float*)d_in[0];
    const float* refp    = (const float*)d_in[1];
    const float* toff    = (const float*)d_in[2];
    const float* inputf  = (const float*)d_in[3];
    const float* W_samp  = (const float*)d_in[4];
    const float* b_samp  = (const float*)d_in[5];
    const float* W_attn  = (const float*)d_in[6];
    const float* b_attn  = (const float*)d_in[7];
    const float* W_tsamp = (const float*)d_in[8];
    const float* b_tsamp = (const float*)d_in[9];
    const float* W_tattn = (const float*)d_in[10];
    const float* b_tattn = (const float*)d_in[11];
    const float* W_val   = (const float*)d_in[12];
    const float* b_val   = (const float*)d_in[13];
    const float* W_out   = (const float*)d_in[14];
    const float* b_out   = (const float*)d_in[15];
    float* out = (float*)d_out;

    // ---- workspace layout ----
    float* ws    = (float*)d_ws;
    float* value = ws;                       // LQ*256
    float* cso   = value + (size_t)LQ * 256; // LQ*256
    float* tso   = cso   + (size_t)LQ * 256; // LQ*256
    float* cawb  = tso   + (size_t)LQ * 256; // LQ*128
    float* tawb  = cawb  + (size_t)LQ * 128; // LQ*128
    float* outh  = tawb  + (size_t)LQ * 128; // LQ*256
    __bf16* wtb  = (__bf16*)(outh + (size_t)LQ * 256);
    __bf16* Wt_val   = wtb;                  // 256*256
    __bf16* Wt_samp  = Wt_val   + 65536;     // 256*256
    __bf16* Wt_tsamp = Wt_samp  + 65536;     // 256*256
    __bf16* Wt_attn  = Wt_tsamp + 65536;     // 128*256
    __bf16* Wt_tattn = Wt_attn  + 32768;     // 128*256
    __bf16* Wt_out   = Wt_tattn + 32768;     // 256*256

    // ---- weight transpose + bf16 convert (tiny) ----
    transpose_convert_w<<<256, 256, 0, stream>>>(W_val,   Wt_val,   256);
    transpose_convert_w<<<256, 256, 0, stream>>>(W_samp,  Wt_samp,  256);
    transpose_convert_w<<<256, 256, 0, stream>>>(W_tsamp, Wt_tsamp, 256);
    transpose_convert_w<<<128, 256, 0, stream>>>(W_attn,  Wt_attn,  128);
    transpose_convert_w<<<128, 256, 0, stream>>>(W_tattn, Wt_tattn, 128);
    transpose_convert_w<<<256, 256, 0, stream>>>(W_out,   Wt_out,   256);

    const dim3 blk(32, 8);
    // tiles = (M/16)*(N/16); blocks = tiles/8 (exact for these shapes)
    gemm_wmma_bf16<<<3060, blk, 0, stream>>>(inputf, Wt_val,   b_val,   value, LQ, 256);
    gemm_wmma_bf16<<<3060, blk, 0, stream>>>(query,  Wt_samp,  b_samp,  cso,   LQ, 256);
    gemm_wmma_bf16<<<3060, blk, 0, stream>>>(query,  Wt_tsamp, b_tsamp, tso,   LQ, 256);
    gemm_wmma_bf16<<<1530, blk, 0, stream>>>(query,  Wt_attn,  b_attn,  cawb,  LQ, 128);
    gemm_wmma_bf16<<<1530, blk, 0, stream>>>(query,  Wt_tattn, b_tattn, tawb,  LQ, 128);

    sample_kernel<<<LQ, blk, 0, stream>>>(value, cawb, tawb, cso, tso, refp, toff, outh);

    gemm_wmma_bf16<<<3060, blk, 0, stream>>>(outh, Wt_out, b_out, out, LQ, 256);
}
